// DGCRM_25443386262431
// MI455X (gfx1250) — matrix operand order repaired and checked by reading, not compiled, not run on hardware
//
#include <hip/hip_runtime.h>
#include <math.h>
#include <stdint.h>

// B=4, N=128, T=96, D=256, G=64, H=4, DK=16
typedef __attribute__((ext_vector_type(16))) _Float16 v16h;
typedef __attribute__((ext_vector_type(4)))  _Float16 v4h;
typedef __attribute__((ext_vector_type(8)))  float    v8f;
// TDM descriptor groups (6-arg toolchain): u32x4, i32x8, i32x4, i32x4, i32x8, i32
typedef __attribute__((__vector_size__(16))) unsigned int tdm4u;
typedef __attribute__((__vector_size__(16))) int          tdm4i;
typedef __attribute__((__vector_size__(32))) int          tdm8i;
// async-to-LDS pointee type per clang diagnostic: vector_size(16) int
typedef __attribute__((__vector_size__(16))) int          vsi4;

#ifdef __has_builtin
#  if __has_builtin(__builtin_amdgcn_global_load_async_to_lds_b128)
#    define HAS_ASYNC_LDS 1
#  endif
#  if __has_builtin(__builtin_amdgcn_tensor_load_to_lds)
#    define HAS_TDM 1
#  endif
#endif
#ifndef HAS_ASYNC_LDS
#define HAS_ASYNC_LDS 0
#endif
#ifndef HAS_TDM
#define HAS_TDM 0
#endif

static __device__ inline v8f vzero8() { v8f z = {}; return z; }
static __device__ inline v4h pack4(float a, float b, float c, float d) {
  v4h p = {(_Float16)a, (_Float16)b, (_Float16)c, (_Float16)d};
  return p;
}

// ---- WMMA fragment helpers (CDNA5 ISA 7.12.2, wave32) ----
static __device__ inline v16h load_a_frag(const _Float16* A, int lda, int m0, int k0, int lane) {
  int m = m0 + (lane & 15);
  int half = lane >> 4;
  const _Float16* ap = A + m * lda + k0 + half * 16;
  v16h a;
#pragma unroll
  for (int i = 0; i < 16; ++i) a[i] = ap[i];
  return a;
}
// B from TRANSPOSED storage Bt[N][K]: 16 contiguous halfwords per lane = 2x ds_load_b128
static __device__ inline v16h load_bt_frag(const _Float16* Bt, int ldk, int k0, int n0, int lane) {
  int n = n0 + (lane & 15);
  int half = lane >> 4;
  const _Float16* bp = Bt + n * ldk + k0 + half * 16;
  v16h b;
#pragma unroll
  for (int i = 0; i < 16; ++i) b[i] = bp[i];
  return b;
}
static __device__ inline v8f wmma32(v16h a, v16h b, v8f c) {
  return __builtin_amdgcn_wmma_f32_16x16x32_f16(false, a, false, b, (short)0, c, false, false);
}

// ---- global(f16) -> LDS panel copy: async-to-LDS when available ----
static __device__ inline void copy_rows_g2l(const _Float16* __restrict__ g, int srcLd,
                                            _Float16* l, int dstLd, int rows, int rowHalves,
                                            int tid) {
#if HAS_ASYNC_LDS
  const int cpr = rowHalves >> 3;  // 16B chunks per row
  const int total = rows * cpr;
  for (int idx = tid; idx < total; idx += 256) {
    int r = idx / cpr, c = idx - r * cpr;
    __attribute__((address_space(1))) vsi4* gp =
        (__attribute__((address_space(1))) vsi4*)(uintptr_t)(g + r * srcLd + c * 8);
    __attribute__((address_space(3))) vsi4* lp =
        (__attribute__((address_space(3))) vsi4*)(uintptr_t)(l + r * dstLd + c * 8);
    __builtin_amdgcn_global_load_async_to_lds_b128(gp, lp, 0, 0);
  }
  asm volatile("s_wait_asynccnt 0" ::: "memory");
#else
  const int cpr = rowHalves >> 1;
  const int total = rows * cpr;
  for (int idx = tid; idx < total; idx += 256) {
    int r = idx / cpr, c = idx - r * cpr;
    ((uint32_t*)(l + r * dstLd))[c] = ((const uint32_t*)(g + r * srcLd))[c];
  }
#endif
}

// ---------------- K0: weight convert + transpose ----------------
__global__ void k0_convT(const float* __restrict__ src, void* dstv, int R, int C, int dstLd, int dstOff) {
  _Float16* dst = (_Float16*)dstv;
  int i = blockIdx.x * blockDim.x + threadIdx.x;
  if (i < R * C) {
    int r = i / C, c = i - r * C;
    dst[c * dstLd + dstOff + r] = (_Float16)src[i];
  }
}
__global__ void k0_permute_qkT(const float* __restrict__ src, void* dstv) {
  _Float16* dst = (_Float16*)dstv;
  int i = blockIdx.x * blockDim.x + threadIdx.x;
  if (i < 4096) {
    int h = i >> 10, g = (i >> 4) & 63, d = i & 15;
    dst[(h * 16 + d) * 64 + g] = (_Float16)src[i];
  }
}

// ---------------- K1: down_dim  (x@Wd1+bd1)@Wd2+bd2 ----------------
__global__ __launch_bounds__(256) void k1_down(const float* __restrict__ x,
                                               const void* Wd1tv, const float* __restrict__ bd1,
                                               const void* Wd2tv, const float* __restrict__ bd2,
                                               float* __restrict__ cur) {
  const _Float16* Wd1t = (const _Float16*)Wd1tv;  // [64][256]
  const _Float16* Wd2t = (const _Float16*)Wd2tv;  // [64][64]
  extern __shared__ char smem[];
  _Float16* xt  = (_Float16*)smem;      // [32][256] 16KB
  _Float16* w1t = xt + 32 * 256;        // [64][256] 32KB
  _Float16* t1  = w1t + 64 * 256;       // [32][64]   4KB
  _Float16* w2t = t1 + 32 * 64;         // [64][64]   8KB
  int tid = threadIdx.x, lane = tid & 31, w = tid >> 5;
  long row0 = (long)blockIdx.x * 32;
  const float4* xv = (const float4*)(x + row0 * 256);
  for (int i = tid; i < 32 * 64; i += 256) {
    float4 v = xv[i];
    *(v4h*)(xt + i * 4) = pack4(v.x, v.y, v.z, v.w);
  }
  copy_rows_g2l(Wd1t, 256, w1t, 256, 64, 256, tid);
  copy_rows_g2l(Wd2t, 64, w2t, 64, 64, 64, tid);
  __syncthreads();
  {  // GEMM1 [32,256]@[256,64] + bd1 -> t1 (f16)
    int m0 = (w >> 2) * 16, n0 = (w & 3) * 16;
    v8f acc = vzero8();
    for (int k0 = 0; k0 < 256; k0 += 32)
      acc = wmma32(load_a_frag(xt, 256, m0, k0, lane), load_bt_frag(w1t, 256, k0, n0, lane), acc);
    int cn = n0 + (lane & 15), half = lane >> 4;
    float bb = bd1[cn];
#pragma unroll
    for (int r = 0; r < 8; ++r) t1[(m0 + half * 8 + r) * 64 + cn] = (_Float16)(acc[r] + bb);
  }
  __syncthreads();
  {  // GEMM2 [32,64]@[64,64] + bd2 -> cur (f32)
    int m0 = (w >> 2) * 16, n0 = (w & 3) * 16;
    v8f acc = vzero8();
    for (int k0 = 0; k0 < 64; k0 += 32)
      acc = wmma32(load_a_frag(t1, 64, m0, k0, lane), load_bt_frag(w2t, 64, k0, n0, lane), acc);
    int cn = n0 + (lane & 15), half = lane >> 4;
    float bb = bd2[cn];
#pragma unroll
    for (int r = 0; r < 8; ++r) cur[(row0 + m0 + half * 8 + r) * 64 + cn] = acc[r] + bb;
  }
}

// ---------------- K2: persistent GRU + dynamic-adjacency graph conv ----------------
static __device__ inline void gate_gemm(const _Float16* acat, _Float16* bW, const _Float16* WgT,
                                        const float* bias, float* out, int act,
                                        int tid, int lane, int w) {
  copy_rows_g2l(WgT, 128, bW, 128, 64, 128, tid);
  __syncthreads();
#pragma unroll
  for (int ti = 0; ti < 4; ++ti) {
    int tIdx = w + ti * 8;
    int m0 = (tIdx >> 2) * 16, n0 = (tIdx & 3) * 16;
    v8f acc = vzero8();
    for (int k0 = 0; k0 < 128; k0 += 32)
      acc = wmma32(load_a_frag(acat, 128, m0, k0, lane), load_bt_frag(bW, 128, k0, n0, lane), acc);
    int cn = n0 + (lane & 15), half = lane >> 4;
    float bb = bias[cn];
#pragma unroll
    for (int r = 0; r < 8; ++r) {
      float v = acc[r] + bb;
      v = act ? tanhf(v) : 1.f / (1.f + __expf(-v));
      out[(m0 + half * 8 + r) * 64 + cn] = v;
    }
  }
  __syncthreads();
}
// h16 (in acat cols 0..63) @ W[64,64] -> per-head zero-padded f16 [4][128][32]
static __device__ inline void qk_gemm_p(const _Float16* acat, _Float16* bW, const _Float16* WmT,
                                        _Float16* qp, int tid, int lane, int w) {
  copy_rows_g2l(WmT, 64, bW, 64, 64, 64, tid);
  __syncthreads();
#pragma unroll
  for (int ti = 0; ti < 4; ++ti) {
    int tIdx = w + ti * 8;
    int m0 = (tIdx >> 2) * 16, n0 = (tIdx & 3) * 16;
    v8f acc = vzero8();
    for (int k0 = 0; k0 < 64; k0 += 32)
      acc = wmma32(load_a_frag(acat, 128, m0, k0, lane), load_bt_frag(bW, 64, k0, n0, lane), acc);
    int cn = n0 + (lane & 15), half = lane >> 4;
    int hd = cn >> 4, d = cn & 15;
#pragma unroll
    for (int r = 0; r < 8; ++r) {
      int row = m0 + half * 8 + r;
      qp[(hd * 128 + row) * 32 + d] = (_Float16)acc[r];
      qp[(hd * 128 + row) * 32 + 16 + d] = (_Float16)0.f;  // K pad 16->32
    }
  }
  __syncthreads();
}

__global__ __launch_bounds__(256) void k2_gru(const float* __restrict__ x,
                                              const float* __restrict__ cur,
                                              const float* __restrict__ h0,
                                              const void* WzCv, const void* WrCv, const void* WhCv,
                                              const void* Wqv, const void* Wkv, const void* Wov,
                                              const float* __restrict__ bz, const float* __restrict__ br,
                                              const float* __restrict__ bh, const float* __restrict__ bo,
                                              float* __restrict__ oh) {
  const _Float16* WzCt = (const _Float16*)WzCv;  // [64][128]
  const _Float16* WrCt = (const _Float16*)WrCv;
  const _Float16* WhCt = (const _Float16*)WhCv;
  const _Float16* WqT  = (const _Float16*)Wqv;   // [64][64]
  const _Float16* WkT  = (const _Float16*)Wkv;
  const _Float16* WoT  = (const _Float16*)Wov;   // [256][256] (N-major)
  extern __shared__ char smem[];
  float*    h32  = (float*)smem;                    // [128][64]  32KB persistent hidden
  _Float16* acat = (_Float16*)(smem + 32 * 1024);   // [128][128] 32KB ([c|h] -> [h_new|rh] -> a16)
  float*    bB   = (float*)(smem + 64 * 1024);      // 32KB: z f32 -> qp f16[4][128][32]
  float*    bC   = (float*)(smem + 96 * 1024);      // 32KB: r/hh f32 -> kp f16[4][128][32]
  char*     bS   = smem + 128 * 1024;               // 64KB: a_avg f32 -> xg16 [128][256]
  char*     bX   = smem + 192 * 1024;               // 68KB: scores f32[128][128]; x_t^T f16[256][136]
  _Float16* bW   = (_Float16*)(smem + 192 * 1024 + 69632);  // 32KB weight staging
  int tid = threadIdx.x, lane = tid & 31, w = tid >> 5;
  int b = blockIdx.x;
  for (int i = tid; i < 128 * 16; i += 256) ((float4*)h32)[i] = ((const float4*)h0)[b * 128 * 16 + i];
  __syncthreads();

  for (int t = 0; t < 96; ++t) {
    // acat = [c_t | h] (f16), float4 loads + packed 8B LDS stores
    for (int i = tid; i < 128 * 16; i += 256) {
      int n = i >> 4, c = i & 15;
      float4 v = ((const float4*)cur)[(((long)(b * 128 + n) * 96 + t) << 4) + c];
      *(v4h*)(acat + n * 128 + c * 4) = pack4(v.x, v.y, v.z, v.w);
      float4 hv = ((const float4*)h32)[i];
      *(v4h*)(acat + n * 128 + 64 + c * 4) = pack4(hv.x, hv.y, hv.z, hv.w);
    }
    __syncthreads();
    gate_gemm(acat, bW, WzCt, bz, bB, 0, tid, lane, w);  // z = sigmoid
    gate_gemm(acat, bW, WrCt, br, bC, 0, tid, lane, w);  // r = sigmoid
    for (int i = tid; i < 128 * 16; i += 256) {          // acat 2nd half <- r*h
      int n = i >> 4, c = i & 15;
      float4 rv = ((const float4*)bC)[i], hv = ((const float4*)h32)[i];
      *(v4h*)(acat + n * 128 + 64 + c * 4) = pack4(rv.x * hv.x, rv.y * hv.y, rv.z * hv.z, rv.w * hv.w);
    }
    __syncthreads();
    gate_gemm(acat, bW, WhCt, bh, bC, 1, tid, lane, w);  // hh = tanh
    for (int i = tid; i < 128 * 16; i += 256) {          // h_new; f16 copy into acat cols 0..63
      int n = i >> 4, c = i & 15;
      float4 z = ((const float4*)bB)[i], hh = ((const float4*)bC)[i], hv = ((const float4*)h32)[i];
      float4 hn;
      hn.x = (1.f - z.x) * hv.x + z.x * hh.x;
      hn.y = (1.f - z.y) * hv.y + z.y * hh.y;
      hn.z = (1.f - z.z) * hv.z + z.z * hh.z;
      hn.w = (1.f - z.w) * hv.w + z.w * hh.w;
      ((float4*)h32)[i] = hn;
      *(v4h*)(acat + n * 128 + c * 4) = pack4(hn.x, hn.y, hn.z, hn.w);
    }
    __syncthreads();
    _Float16* qp = (_Float16*)bB;  // [4][128][32] zero-padded
    _Float16* kp = (_Float16*)bC;
    qk_gemm_p(acat, bW, WqT, qp, tid, lane, w);
    qk_gemm_p(acat, bW, WkT, kp, tid, lane, w);

    // scores via WMMA (K padded to 32): per head 64 tiles of q_h[128,32] @ k_h^T
    float* aavg = (float*)bS;
    float* sc   = (float*)bX;
    for (int hd = 0; hd < 4; ++hd) {
      const _Float16* qh = qp + hd * 128 * 32;
      const _Float16* kh = kp + hd * 128 * 32;
#pragma unroll
      for (int ti = 0; ti < 8; ++ti) {
        int tIdx = w + ti * 8;
        int m0 = (tIdx >> 3) * 16, n0 = (tIdx & 7) * 16;
        v8f acc = wmma32(load_a_frag(qh, 32, m0, 0, lane), load_bt_frag(kh, 32, 0, n0, lane), vzero8());
        int cn = n0 + (lane & 15), half = lane >> 4;
#pragma unroll
        for (int r = 0; r < 8; ++r) sc[(m0 + half * 8 + r) * 128 + cn] = acc[r] * 0.25f;  // /sqrt(16)
      }
      __syncthreads();
      if (tid < 128) {
        int n = tid;
        float mx = -1e30f;
        for (int m = 0; m < 128; ++m) mx = fmaxf(mx, sc[n * 128 + m]);
        float sum = 0.f;
        for (int m = 0; m < 128; ++m) { float e = __expf(sc[n * 128 + m] - mx); sum += e; sc[n * 128 + m] = e; }
        float inv = 0.25f / sum;  // 0.25 = mean over H=4 heads
        for (int m = 0; m < 128; ++m) {
          float v = sc[n * 128 + m] * inv;
          if (hd == 0) aavg[n * 128 + m] = v; else aavg[n * 128 + m] += v;
        }
      }
      __syncthreads();
    }
    // a16 <- aavg ; x_t -> transposed f16 [d=256][n=128] ld 136
    _Float16* a16 = acat;
    _Float16* xtT = (_Float16*)bX;
    for (int i = tid; i < 128 * 32; i += 256) {
      float4 v = ((const float4*)aavg)[i];
      *(v4h*)(a16 + i * 4) = pack4(v.x, v.y, v.z, v.w);
    }
    for (int i = tid; i < 128 * 64; i += 256) {
      int n = i >> 6, dq = i & 63;
      float4 v = ((const float4*)x)[(((long)(b * 128 + n) * 96 + t) << 6) + dq];
      int d = dq * 4;
      xtT[d * 136 + n] = (_Float16)v.x;
      xtT[(d + 1) * 136 + n] = (_Float16)v.y;
      xtT[(d + 2) * 136 + n] = (_Float16)v.z;
      xtT[(d + 3) * 136 + n] = (_Float16)v.w;
    }
    __syncthreads();
    // xg = a_avg @ x_t : [128,128]@[128,256] -> xg16
    _Float16* xg16 = (_Float16*)bS;
#pragma unroll
    for (int ti = 0; ti < 16; ++ti) {
      int tIdx = w + ti * 8;
      int m0 = (tIdx >> 4) * 16, n0 = (tIdx & 15) * 16;
      v8f acc = vzero8();
      for (int k0 = 0; k0 < 128; k0 += 32)
        acc = wmma32(load_a_frag(a16, 128, m0, k0, lane), load_bt_frag(xtT, 136, k0, n0, lane), acc);
      int cn = n0 + (lane & 15), half = lane >> 4;
#pragma unroll
      for (int r = 0; r < 8; ++r) xg16[(m0 + half * 8 + r) * 256 + cn] = (_Float16)acc[r];
    }
    __syncthreads();
    // out = x_t + xg @ Wo + bo : Wo^T K-panels via TDM / async
    {
      v8f acc[16];
#pragma unroll
      for (int i = 0; i < 16; ++i) acc[i] = vzero8();
      for (int ko = 0; ko < 256; ko += 64) {
#if HAS_TDM
        if (w == 0) {
          unsigned long long ga = (unsigned long long)(uintptr_t)(WoT + ko);
          tdm4u g0 = {};
          g0[0] = 1u;                                     // count=1
          g0[1] = (unsigned)(uintptr_t)bW;                // lds_addr
          g0[2] = (unsigned)ga;                           // global_addr lo
          g0[3] = ((unsigned)(ga >> 32) & 0x1FFFFFFu) | (2u << 30);  // addr hi | type=2
          tdm8i g1 = {};
          g1[0] = (2 << 16);          // data_size = 4B
          g1[1] = (int)(128u << 16);  // tensor_dim0 = 128 dwords
          g1[2] = (int)(256u << 16);  // tensor_dim1 = 256
          g1[3] = (int)(32u << 16);   // tile_dim0 = 32 dwords
          g1[4] = 256;                // tile_dim1 = 256
          g1[5] = 128;                // tensor_dim0_stride = 128 dwords
          tdm4i z4 = {};
          tdm8i z8 = {};
          __builtin_amdgcn_tensor_load_to_lds(g0, g1, z4, z4, z8, 0);
        }
        __builtin_amdgcn_s_wait_tensorcnt((short)0);
#else
        copy_rows_g2l(WoT + ko, 256, bW, 64, 256, 64, tid);
#endif
        __syncthreads();
#pragma unroll
        for (int ti = 0; ti < 16; ++ti) {
          int tIdx = w + ti * 8;
          int m0 = (tIdx >> 4) * 16, n0 = (tIdx & 15) * 16;
          for (int k0 = 0; k0 < 64; k0 += 32)
            acc[ti] = wmma32(load_a_frag(xg16, 256, m0, ko + k0, lane),
                             load_bt_frag(bW, 64, k0, n0, lane), acc[ti]);
        }
        __syncthreads();
      }
#pragma unroll
      for (int ti = 0; ti < 16; ++ti) {
        int tIdx = w + ti * 8;
        int m0 = (tIdx >> 4) * 16, n0 = (tIdx & 15) * 16;
        int cn = n0 + (lane & 15), half = lane >> 4;
        float bb = bo[cn];
#pragma unroll
        for (int r = 0; r < 8; ++r) {
          int nn = m0 + half * 8 + r;
          long xi = ((long)(b * 128 + nn) * 96 + t) * 256 + cn;
          oh[xi] = x[xi] + acc[ti][r] + bb;
        }
      }
    }
    __syncthreads();
  }
}

// ---------------- K3: 2*oh -> LN1 -> MLP(GELU) -> +res -> LN2 ----------------
__global__ __launch_bounds__(256) void k3_mlp(const float* __restrict__ ohin,
                                              const void* Wm1tv, const float* __restrict__ bm1,
                                              const void* Wm2tv, const float* __restrict__ bm2,
                                              const float* __restrict__ g1, const float* __restrict__ b1,
                                              const float* __restrict__ g2, const float* __restrict__ b2,
                                              float* __restrict__ outp) {
  const _Float16* Wm1t = (const _Float16*)Wm1tv;  // [512][256]
  const _Float16* Wm2t = (const _Float16*)Wm2tv;  // [256][512]
  extern __shared__ char smem[];
  float*    u32  = (float*)smem;                    // [32][256] 32KB
  _Float16* u16  = (_Float16*)(smem + 32 * 1024);   // [32][256] 16KB
  _Float16* m16  = (_Float16*)(smem + 48 * 1024);   // [32][512] 32KB
  _Float16* wst  = (_Float16*)(smem + 80 * 1024);   // 64KB panels (max [512][64])
  float*    rowm = (float*)(smem + 144 * 1024);
  float*    rowr = rowm + 32;
  int tid = threadIdx.x, lane = tid & 31, w = tid >> 5;
  long row0 = (long)blockIdx.x * 32;
  const float4* ov = (const float4*)(ohin + row0 * 256);
  for (int i = tid; i < 32 * 64; i += 256) {
    float4 v = ov[i];
    v.x *= 2.f; v.y *= 2.f; v.z *= 2.f; v.w *= 2.f;
    ((float4*)u32)[i] = v;
  }
  __syncthreads();
  if (tid < 32) {  // LN1
    float m = 0.f;
    for (int c = 0; c < 256; ++c) m += u32[tid * 256 + c];
    m *= (1.f / 256.f);
    float v = 0.f;
    for (int c = 0; c < 256; ++c) { float d = u32[tid * 256 + c] - m; v += d * d; }
    v *= (1.f / 256.f);
    rowm[tid] = m; rowr[tid] = rsqrtf(v + 1e-5f);
  }
  __syncthreads();
  for (int i = tid; i < 32 * 256; i += 256) {
    int r = i >> 8, c = i & 255;
    float uv = (u32[i] - rowm[r]) * rowr[r] * g1[c] + b1[c];
    u32[i] = uv; u16[i] = (_Float16)uv;
  }
  __syncthreads();
  {  // MLP1 [32,256]@[256,512] + bm1, GELU(tanh) -> m16
    v8f acc[8];
#pragma unroll
    for (int i = 0; i < 8; ++i) acc[i] = vzero8();
    for (int ko = 0; ko < 256; ko += 64) {
      copy_rows_g2l(Wm1t + ko, 256, wst, 64, 512, 64, tid);
      __syncthreads();
#pragma unroll
      for (int ti = 0; ti < 8; ++ti) {
        int tIdx = w + ti * 8;
        int m0 = (tIdx >> 5) * 16, n0 = (tIdx & 31) * 16;
        for (int k0 = 0; k0 < 64; k0 += 32)
          acc[ti] = wmma32(load_a_frag(u16, 256, m0, ko + k0, lane),
                           load_bt_frag(wst, 64, k0, n0, lane), acc[ti]);
      }
      __syncthreads();
    }
#pragma unroll
    for (int ti = 0; ti < 8; ++ti) {
      int tIdx = w + ti * 8;
      int m0 = (tIdx >> 5) * 16, n0 = (tIdx & 31) * 16;
      int cn = n0 + (lane & 15), half = lane >> 4;
      float bb = bm1[cn];
#pragma unroll
      for (int r = 0; r < 8; ++r) {
        float vv = acc[ti][r] + bb;
        float gv = 0.5f * vv * (1.f + tanhf(0.7978845608f * (vv + 0.044715f * vv * vv * vv)));
        m16[(m0 + half * 8 + r) * 512 + cn] = (_Float16)gv;
      }
    }
  }
  __syncthreads();
  {  // MLP2 [32,512]@[512,256] + bm2, residual into u32
    v8f acc[4];
#pragma unroll
    for (int i = 0; i < 4; ++i) acc[i] = vzero8();
    for (int ko = 0; ko < 512; ko += 64) {
      copy_rows_g2l(Wm2t + ko, 512, wst, 64, 256, 64, tid);
      __syncthreads();
#pragma unroll
      for (int ti = 0; ti < 4; ++ti) {
        int tIdx = w + ti * 8;
        int m0 = (tIdx >> 2) * 16, n0 = (tIdx & 3) * 16;
        for (int k0 = 0; k0 < 64; k0 += 32)
          acc[ti] = wmma32(load_a_frag(m16, 512, m0, ko + k0, lane),
                           load_bt_frag(wst, 64, k0, n0, lane), acc[ti]);
      }
      __syncthreads();
    }
#pragma unroll
    for (int ti = 0; ti < 4; ++ti) {
      int tIdx = w + ti * 8;
      int m0 = (tIdx >> 2) * 16, n0 = (tIdx & 3) * 16;
      int cn = n0 + (lane & 15), half = lane >> 4;
      float bb = bm2[cn];
#pragma unroll
      for (int r = 0; r < 8; ++r) u32[(m0 + half * 8 + r) * 256 + cn] += acc[ti][r] + bb;
    }
  }
  __syncthreads();
  if (tid < 32) {  // LN2
    float m = 0.f;
    for (int c = 0; c < 256; ++c) m += u32[tid * 256 + c];
    m *= (1.f / 256.f);
    float v = 0.f;
    for (int c = 0; c < 256; ++c) { float d = u32[tid * 256 + c] - m; v += d * d; }
    v *= (1.f / 256.f);
    rowm[tid] = m; rowr[tid] = rsqrtf(v + 1e-5f);
  }
  __syncthreads();
  for (int i = tid; i < 32 * 256; i += 256) {
    int r = i >> 8, c = i & 255;
    outp[row0 * 256 + i] = (u32[i] - rowm[r]) * rowr[r] * g2[c] + b2[c];
  }
}

// ---------------- host ----------------
extern "C" void kernel_launch(void* const* d_in, const int* in_sizes, int n_in,
                              void* d_out, int out_size, void* d_ws, size_t ws_size,
                              hipStream_t stream) {
  const float* x   = (const float*)d_in[0];
  const float* h0  = (const float*)d_in[1];
  const float* Wd1 = (const float*)d_in[2];
  const float* bd1 = (const float*)d_in[3];
  const float* Wd2 = (const float*)d_in[4];
  const float* bd2 = (const float*)d_in[5];
  const float* Wxz = (const float*)d_in[6];
  const float* Whz = (const float*)d_in[7];
  const float* bz  = (const float*)d_in[8];
  const float* Wxr = (const float*)d_in[9];
  const float* Whr = (const float*)d_in[10];
  const float* br  = (const float*)d_in[11];
  const float* Wxh = (const float*)d_in[12];
  const float* Whh = (const float*)d_in[13];
  const float* bh  = (const float*)d_in[14];
  const float* Wq  = (const float*)d_in[15];
  const float* Wk  = (const float*)d_in[16];
  const float* Wo  = (const float*)d_in[17];
  const float* bo  = (const float*)d_in[18];
  const float* g1  = (const float*)d_in[19];
  const float* b1  = (const float*)d_in[20];
  const float* g2  = (const float*)d_in[21];
  const float* b2  = (const float*)d_in[22];
  const float* Wm1 = (const float*)d_in[23];
  const float* bm1 = (const float*)d_in[24];
  const float* Wm2 = (const float*)d_in[25];
  const float* bm2 = (const float*)d_in[26];

  char* ws = (char*)d_ws;
  float* cur = (float*)ws;               // [4*128*96*64]  f32 = 12,582,912 B
  float* oh  = (float*)(ws + 12582912);  // [4*128*96*256] f32 = 50,331,648 B
  char* f16b = ws + 62914560;            // f16 transposed-weight arena
  void* Wd1h = (void*)(f16b + 0);
  void* Wd2h = (void*)(f16b + 32768);
  void* WzCh = (void*)(f16b + 40960);
  void* WrCh = (void*)(f16b + 57344);
  void* WhCh = (void*)(f16b + 73728);
  void* WqAh = (void*)(f16b + 90112);
  void* WkAh = (void*)(f16b + 98304);
  void* Wo16 = (void*)(f16b + 106496);
  void* Wm1h = (void*)(f16b + 237568);
  void* Wm2h = (void*)(f16b + 499712);

  k0_convT<<<64, 256, 0, stream>>>(Wd1, Wd1h, 256, 64, 256, 0);
  k0_convT<<<16, 256, 0, stream>>>(Wd2, Wd2h, 64, 64, 64, 0);
  k0_convT<<<16, 256, 0, stream>>>(Wxz, WzCh, 64, 64, 128, 0);
  k0_convT<<<16, 256, 0, stream>>>(Whz, WzCh, 64, 64, 128, 64);
  k0_convT<<<16, 256, 0, stream>>>(Wxr, WrCh, 64, 64, 128, 0);
  k0_convT<<<16, 256, 0, stream>>>(Whr, WrCh, 64, 64, 128, 64);
  k0_convT<<<16, 256, 0, stream>>>(Wxh, WhCh, 64, 64, 128, 0);
  k0_convT<<<16, 256, 0, stream>>>(Whh, WhCh, 64, 64, 128, 64);
  k0_permute_qkT<<<16, 256, 0, stream>>>(Wq, WqAh);
  k0_permute_qkT<<<16, 256, 0, stream>>>(Wk, WkAh);
  k0_convT<<<256, 256, 0, stream>>>(Wo, Wo16, 256, 256, 256, 0);
  k0_convT<<<512, 256, 0, stream>>>(Wm1, Wm1h, 256, 512, 256, 0);
  k0_convT<<<512, 256, 0, stream>>>(Wm2, Wm2h, 512, 256, 512, 0);

  k1_down<<<1536, 256, 61440, stream>>>(x, Wd1h, bd1, Wd2h, bd2, cur);
  k2_gru<<<4, 256, 299008, stream>>>(x, cur, h0, WzCh, WrCh, WhCh, WqAh, WkAh, Wo16,
                                     bz, br, bh, bo, oh);
  k3_mlp<<<1536, 256, 147712, stream>>>(oh, Wm1h, bm1, Wm2h, bm2, g1, b1, g2, b2, (float*)d_out);
}
